// ContrastiveLearningModule_70909910057292
// MI455X (gfx1250) — matrix operand order repaired and checked by reading, not compiled, not run on hardware
//
#include <hip/hip_runtime.h>
#include <hip/hip_bf16.h>
#include <math.h>

// ---------------------------------------------------------------------------
// ContrastiveLearning loss for N=4096, D=2048, T=0.5, fp32 inputs.
// loss = 2 + log( sum_ij exp(2*simneg_ij - 2) ) - 2 * dot(colsum(A_n),colsum(B_n)) / N^2
// ---------------------------------------------------------------------------

#define N_ROWS 4096
#define D_COLS 2048

typedef __attribute__((ext_vector_type(8)))  __bf16 v8bf;
typedef __attribute__((ext_vector_type(16))) __bf16 v16bf;
typedef __attribute__((ext_vector_type(8)))  float  v8f;

// ---------------------------------------------------------------------------
// Kernel 1: per-row L2 normalization.
//   blockIdx.y = 0 -> x_source      -> bf16 normalized rows (Abf)
//   blockIdx.y = 1 -> x_raw_target  -> bf16 normalized rows (Cbf)
//   blockIdx.y = 2 -> x_bc_target   -> only 1/||row|| (rnormB)
// 256 threads/block, 8 elems/thread.
// ---------------------------------------------------------------------------
__global__ void __launch_bounds__(256)
rownorm_kernel(const float* __restrict__ A, const float* __restrict__ B,
               const float* __restrict__ C,
               __bf16* __restrict__ Abf, __bf16* __restrict__ Cbf,
               float* __restrict__ rnormB) {
    const int r     = blockIdx.x;
    const int which = blockIdx.y;
    const float* src = (which == 0) ? A : ((which == 1) ? C : B);
    const float* row = src + (size_t)r * D_COLS;

    const int t = threadIdx.x;
    float4 x0 = ((const float4*)row)[t * 2 + 0];
    float4 x1 = ((const float4*)row)[t * 2 + 1];
    float ss = x0.x * x0.x + x0.y * x0.y + x0.z * x0.z + x0.w * x0.w +
               x1.x * x1.x + x1.y * x1.y + x1.z * x1.z + x1.w * x1.w;

    __shared__ float red[256];
    red[t] = ss;
    __syncthreads();
    for (int off = 128; off > 0; off >>= 1) {
        if (t < off) red[t] += red[t + off];
        __syncthreads();
    }
    const float rn = rsqrtf(red[0]);

    if (which == 2) {
        if (t == 0) rnormB[r] = rn;
        return;
    }

    __bf16* dst = ((which == 0) ? Abf : Cbf) + (size_t)r * D_COLS + t * 8;
    v8bf o;
    o[0] = (__bf16)(x0.x * rn); o[1] = (__bf16)(x0.y * rn);
    o[2] = (__bf16)(x0.z * rn); o[3] = (__bf16)(x0.w * rn);
    o[4] = (__bf16)(x1.x * rn); o[5] = (__bf16)(x1.y * rn);
    o[6] = (__bf16)(x1.z * rn); o[7] = (__bf16)(x1.w * rn);
    *(v8bf*)dst = o;
}

// ---------------------------------------------------------------------------
// Kernel 2: column sums of normalized A (from bf16) and normalized B
// (f32 * per-row rnorm).  2048 threads total (grid 8 x 256). Coalesced:
// adjacent threads read adjacent columns. Also zeroes the exp accumulator.
// ---------------------------------------------------------------------------
__global__ void __launch_bounds__(256)
colsum_kernel(const __bf16* __restrict__ Abf, const float* __restrict__ B,
              const float* __restrict__ rnormB,
              float* __restrict__ colA, float* __restrict__ colB,
              float* __restrict__ Eacc) {
    const int d = blockIdx.x * 256 + threadIdx.x;   // 0..2047
    float sA = 0.f, sB = 0.f;
    for (int r = 0; r < N_ROWS; ++r) {
        sA += (float)Abf[(size_t)r * D_COLS + d];
        sB += B[(size_t)r * D_COLS + d] * rnormB[r];
    }
    colA[d] = sA;
    colB[d] = sB;
    if (d == 0) *Eacc = 0.0f;
}

// ---------------------------------------------------------------------------
// Kernel 3: fused  sum_ij exp(2 * (A_n C_n^T)_ij - 2)  via bf16 WMMA.
//   Block = 256 threads = 8 waves, block tile 128(M) x 256(N).
//   Wave tile 64x64 = 4x4 fragments of v_wmma_f32_16x16x32_bf16.
//   K loop: 2048/32 = 64 iterations, 16 WMMA per iteration per wave.
// ---------------------------------------------------------------------------
__global__ void __launch_bounds__(256)
simexp_gemm_kernel(const __bf16* __restrict__ Abf,
                   const __bf16* __restrict__ Cbf,
                   float* __restrict__ Eacc) {
    const int tid  = threadIdx.x;
    const int wave = tid >> 5;
    const int lane = tid & 31;
    const int lr   = lane & 15;
    const int khA  = (lane >> 4) * 16;   // A frag: two 8-elem runs at khA, khA+8
    const int khB  = (lane >> 4) * 16;   // B frag: one 16-elem run at khB

    const int m0 = blockIdx.y * 128 + (wave >> 2) * 64;
    const int n0 = blockIdx.x * 256 + (wave & 3) * 64;

    const __bf16* pa[4];
    const __bf16* pb[4];
#pragma unroll
    for (int i = 0; i < 4; ++i) {
        pa[i] = Abf + (size_t)(m0 + 16 * i + lr) * D_COLS + khA;
        pb[i] = Cbf + (size_t)(n0 + 16 * i + lr) * D_COLS + khB;
    }

    v8f acc[4][4];
    const v8f vzero = {0.f, 0.f, 0.f, 0.f, 0.f, 0.f, 0.f, 0.f};
#pragma unroll
    for (int i = 0; i < 4; ++i)
#pragma unroll
        for (int j = 0; j < 4; ++j) acc[i][j] = vzero;

    for (int k0 = 0; k0 < D_COLS; k0 += 32) {
        v16bf a[4], b[4];
#pragma unroll
        for (int i = 0; i < 4; ++i) {
            v8bf lo = *(const v8bf*)(pa[i] + k0);
            v8bf hi = *(const v8bf*)(pa[i] + k0 + 8);
            a[i] = __builtin_shufflevector(lo, hi, 0, 1, 2, 3, 4, 5, 6, 7,
                                           8, 9, 10, 11, 12, 13, 14, 15);
        }
#pragma unroll
        for (int j = 0; j < 4; ++j) {
            b[j] = *(const v16bf*)(pb[j] + k0);
        }
#pragma unroll
        for (int i = 0; i < 4; ++i)
#pragma unroll
            for (int j = 0; j < 4; ++j) {
                acc[i][j] = __builtin_amdgcn_wmma_f32_16x16x32_bf16(
                    /*neg_a=*/false, a[i], /*neg_b=*/false, b[j],
                    /*c_mod=*/(short)0, acc[i][j],
                    /*reuse_a=*/false, /*reuse_b=*/false);
            }
    }

    // Epilogue: local sum of exp(2*s - 2) over this lane's 128 outputs.
    float local = 0.f;
#pragma unroll
    for (int i = 0; i < 4; ++i)
#pragma unroll
        for (int j = 0; j < 4; ++j)
#pragma unroll
            for (int e = 0; e < 8; ++e)
                local += __expf(2.0f * acc[i][j][e] - 2.0f);

    __shared__ float red[256];
    red[tid] = local;
    __syncthreads();
    for (int off = 128; off > 0; off >>= 1) {
        if (tid < off) red[tid] += red[tid + off];
        __syncthreads();
    }
    if (tid == 0) atomicAdd(Eacc, red[0]);
}

// ---------------------------------------------------------------------------
// Kernel 4: finalize.  P = dot(colA, colB);  loss = 2 + log(E) - 2P/N^2.
// ---------------------------------------------------------------------------
__global__ void __launch_bounds__(256)
finalize_kernel(const float* __restrict__ colA, const float* __restrict__ colB,
                const float* __restrict__ Eacc, float* __restrict__ out) {
    const int t = threadIdx.x;
    float p = 0.f;
    for (int d = t; d < D_COLS; d += 256) p += colA[d] * colB[d];
    __shared__ float red[256];
    red[t] = p;
    __syncthreads();
    for (int off = 128; off > 0; off >>= 1) {
        if (t < off) red[t] += red[t + off];
        __syncthreads();
    }
    if (t == 0) {
        const float invNN = 1.0f / ((float)N_ROWS * (float)N_ROWS);
        out[0] = 2.0f + logf(*Eacc) - 2.0f * red[0] * invNN;
    }
}

// ---------------------------------------------------------------------------
extern "C" void kernel_launch(void* const* d_in, const int* in_sizes, int n_in,
                              void* d_out, int out_size, void* d_ws, size_t ws_size,
                              hipStream_t stream) {
    const float* A = (const float*)d_in[0];  // x_source
    const float* B = (const float*)d_in[1];  // x_bc_target
    const float* C = (const float*)d_in[2];  // x_raw_target

    char* ws = (char*)d_ws;
    const size_t bfBytes = (size_t)N_ROWS * D_COLS * sizeof(__bf16);  // 16 MB
    __bf16* Abf   = (__bf16*)(ws);
    __bf16* Cbf   = (__bf16*)(ws + bfBytes);
    float* rnormB = (float*)(ws + 2 * bfBytes);
    float* colA   = (float*)(ws + 2 * bfBytes + N_ROWS * sizeof(float));
    float* colB   = (float*)(ws + 2 * bfBytes + N_ROWS * sizeof(float) + D_COLS * sizeof(float));
    float* Eacc   = (float*)(ws + 2 * bfBytes + N_ROWS * sizeof(float) + 2 * D_COLS * sizeof(float));

    // 1) row norms / bf16 conversion
    rownorm_kernel<<<dim3(N_ROWS, 3), 256, 0, stream>>>(A, B, C, Abf, Cbf, rnormB);
    // 2) column sums + zero accumulator
    colsum_kernel<<<dim3(D_COLS / 256), 256, 0, stream>>>(Abf, B, rnormB, colA, colB, Eacc);
    // 3) fused WMMA GEMM + exp-sum   (grid: N/256 x M/128)
    simexp_gemm_kernel<<<dim3(N_ROWS / 256, N_ROWS / 128), 256, 0, stream>>>(Abf, Cbf, Eacc);
    // 4) scalar finalize
    finalize_kernel<<<1, 256, 0, stream>>>(colA, colB, Eacc, (float*)d_out);
}
